// Fusion_17669495456333
// MI455X (gfx1250) — matrix operand order repaired and checked by reading, not compiled, not run on hardware
//
#include <hip/hip_runtime.h>
#include <hip/hip_bf16.h>

// MI455X / gfx1250 implementation. HBM-bound (~350-400MB traffic @23.3TB/s);
// GEMMs done with v_wmma_f32_16x16x32_bf16 (bf16 in, f32 accumulate).
// Workspace layout (needs ~68MB):
//   [0, 64MiB)              search_c  f32  (64*1024*256)
//   +67108864               W         f32  (64*1024)
//   +262144                 ftmean    f32  (64*256)
//   +65536                  wmin[64], wmax[64], pos[64] (int)

typedef __attribute__((ext_vector_type(16))) __bf16 v16bf;
typedef __attribute__((ext_vector_type(8)))  float  v8f;

#define B_   64
#define L_   1152
#define D_   256
#define TF_  128
#define S_   1024
#define SCALE_ 0.0625f   // D^-0.5 = 1/16

__device__ __forceinline__ float4 ld4(const float* p) { return *(const float4*)p; }
__device__ __forceinline__ float4 sub4(float4 a, float4 b) {
  return make_float4(a.x - b.x, a.y - b.y, a.z - b.z, a.w - b.w);
}
__device__ __forceinline__ void pack8(v16bf& v, int base, float4 a, float4 b) {
  v[base + 0] = (__bf16)a.x; v[base + 1] = (__bf16)a.y;
  v[base + 2] = (__bf16)a.z; v[base + 3] = (__bf16)a.w;
  v[base + 4] = (__bf16)b.x; v[base + 5] = (__bf16)b.y;
  v[base + 6] = (__bf16)b.z; v[base + 7] = (__bf16)b.w;
}
__device__ __forceinline__ v8f bfmma(v16bf a, v16bf b, v8f c) {
  // 8 args: (neg_a, A, neg_b, B, c_mod, C, reuse_a, reuse_b)
  return __builtin_amdgcn_wmma_f32_16x16x32_bf16(false, a, false, b, (short)0, c,
                                                 false, false);
}

__global__ void k0_init(int* __restrict__ pos) {
  if (threadIdx.x < B_) pos[threadIdx.x] = 0;
}

// ---------------------------------------------------------------------------
// Kernel 1: pos[b] = #{ (t,s) : ft[b,t,:] . (x-xi)[b,128+s,:] > 0 }
// One 16x16 WMMA tile per wave; 8 waves/block; grid = 64 b * 8 ttile * 8 sgrp.
// ---------------------------------------------------------------------------
__global__ void __launch_bounds__(256) k1_count(const float* __restrict__ x,
                                                const float* __restrict__ xi,
                                                const float* __restrict__ ft,
                                                int* __restrict__ pos) {
  const int wave = threadIdx.x >> 5, lane = threadIdx.x & 31;
  const int bid = blockIdx.x;          // 64 * 64 blocks
  const int b   = bid >> 6;
  const int rem = bid & 63;
  const int t0  = (rem >> 3) << 4;                 // t-tile * 16
  const int s0  = (((rem & 7) << 3) + wave) << 4;  // s-tile * 16
  const int m   = lane & 15;
  const int kbA = (lane < 16) ? 0 : 16;
  const int kbB = (lane < 16) ? 0 : 16;

  const float* arow  = ft + ((size_t)b * TF_ + t0 + m) * D_;
  const float* xrow  = x  + ((size_t)b * L_ + TF_ + s0 + m) * D_;
  const float* xirow = xi + ((size_t)b * L_ + TF_ + s0 + m) * D_;

  v8f c = {};
  #pragma unroll 2
  for (int k0 = 0; k0 < D_; k0 += 32) {
    v16bf av, bv;
    pack8(av, 0, ld4(arow + k0 + kbA),      ld4(arow + k0 + kbA + 4));
    pack8(av, 8, ld4(arow + k0 + kbA + 8), ld4(arow + k0 + kbA + 12));
    float4 p0 = ld4(xrow + k0 + kbB),       p1 = ld4(xrow + k0 + kbB + 4);
    float4 p2 = ld4(xrow + k0 + kbB + 8),   p3 = ld4(xrow + k0 + kbB + 12);
    float4 q0 = ld4(xirow + k0 + kbB),      q1 = ld4(xirow + k0 + kbB + 4);
    float4 q2 = ld4(xirow + k0 + kbB + 8),  q3 = ld4(xirow + k0 + kbB + 12);
    pack8(bv, 0, sub4(p0, q0), sub4(p1, q1));
    pack8(bv, 8, sub4(p2, q2), sub4(p3, q3));
    c = bfmma(av, bv, c);
  }

  int cnt = 0;
  #pragma unroll
  for (int i = 0; i < 8; i++) cnt += (c[i] > 0.0f) ? 1 : 0;
  for (int off = 16; off > 0; off >>= 1) cnt += __shfl_down(cnt, off, 32);
  if (lane == 0) atomicAdd(&pos[b], cnt);
}

// ---------------------------------------------------------------------------
// Kernel 2: search_c[r,d] = sum_k cat[r,k]*conv_w[d,k] + conv_b[d]
//   r = b*1024+s (M=65536), d (N=256), k (K=512, first 256 from x, rest xi).
// One block per M-tile; each wave owns two N-tiles (A reused for both).
// ---------------------------------------------------------------------------
__global__ void __launch_bounds__(256) k2_conv(const float* __restrict__ x,
                                               const float* __restrict__ xi,
                                               const float* __restrict__ cw,
                                               const float* __restrict__ cb,
                                               float* __restrict__ sc) {
  const int wave = threadIdx.x >> 5, lane = threadIdx.x & 31;
  const int mt = blockIdx.x;                 // 4096 M-tiles
  const int m  = lane & 15;
  const int r  = mt * 16 + m;
  const int b  = r >> 10, s = r & 1023;
  const int kbA = (lane < 16) ? 0 : 16;
  const int kbB = (lane < 16) ? 0 : 16;
  const int d0 = wave * 16 + m;
  const int d1 = (wave + 8) * 16 + m;

  const float* xrow  = x  + ((size_t)b * L_ + TF_ + s) * D_;
  const float* xirow = xi + ((size_t)b * L_ + TF_ + s) * D_;
  const float* w0 = cw + (size_t)d0 * (2 * D_);
  const float* w1 = cw + (size_t)d1 * (2 * D_);

  v8f c0 = {}, c1 = {};
  for (int k0 = 0; k0 < 2 * D_; k0 += 32) {
    const float* src = (k0 < D_) ? (xrow + k0) : (xirow + (k0 - D_));
    v16bf av, b0v, b1v;
    pack8(av, 0, ld4(src + kbA),      ld4(src + kbA + 4));
    pack8(av, 8, ld4(src + kbA + 8), ld4(src + kbA + 12));
    pack8(b0v, 0, ld4(w0 + k0 + kbB),     ld4(w0 + k0 + kbB + 4));
    pack8(b0v, 8, ld4(w0 + k0 + kbB + 8), ld4(w0 + k0 + kbB + 12));
    pack8(b1v, 0, ld4(w1 + k0 + kbB),     ld4(w1 + k0 + kbB + 4));
    pack8(b1v, 8, ld4(w1 + k0 + kbB + 8), ld4(w1 + k0 + kbB + 12));
    c0 = bfmma(av, b0v, c0);
    c1 = bfmma(av, b1v, c1);
  }

  const float bias0 = cb[d0], bias1 = cb[d1];
  const int mofs = (lane < 16) ? 0 : 8;
  #pragma unroll
  for (int i = 0; i < 8; i++) {
    const size_t rr = (size_t)(mt * 16 + i + mofs);
    sc[rr * D_ + d0] = c0[i] + bias0;
    sc[rr * D_ + d1] = c1[i] + bias1;
  }
}

// ---------------------------------------------------------------------------
// Kernel 3a: ftmean[b,d] = mean_t ft[b,t,d]
// ---------------------------------------------------------------------------
__global__ void __launch_bounds__(256) k3a_ftmean(const float* __restrict__ ft,
                                                  float* __restrict__ fm) {
  const int b = blockIdx.x, d = threadIdx.x;
  float s = 0.f;
  for (int t = 0; t < TF_; t++) s += ft[((size_t)b * TF_ + t) * D_ + d];
  fm[b * D_ + d] = s * (1.0f / TF_);
}

// Kernel 3b: W[b,s] = exp(softplus(mean_t attn)) = 1 + exp(scale * fm.sc_row)
__global__ void __launch_bounds__(256) k3b_W(const float* __restrict__ sc,
                                             const float* __restrict__ fm,
                                             float* __restrict__ W) {
  const int idx = blockIdx.x * 256 + threadIdx.x;  // 0..65535
  const int b = idx >> 10;
  const float* f = fm + (size_t)b * D_;
  const float* row = sc + (size_t)idx * D_;
  float acc = 0.f;
  #pragma unroll 4
  for (int d = 0; d < D_; d++) acc += f[d] * row[d];
  W[idx] = 1.0f + __expf(acc * SCALE_);
}

// Kernel 3c: per-batch min/max of W over S
__global__ void __launch_bounds__(256) k3c_minmax(const float* __restrict__ W,
                                                  float* __restrict__ wmn,
                                                  float* __restrict__ wmx) {
  __shared__ float smn[256], smx[256];
  const int b = blockIdx.x, t = threadIdx.x;
  float mn = 3.4e38f, mx = -3.4e38f;
  for (int s = t; s < S_; s += 256) {
    const float v = W[b * S_ + s];
    mn = fminf(mn, v); mx = fmaxf(mx, v);
  }
  smn[t] = mn; smx[t] = mx;
  __syncthreads();
  for (int o = 128; o > 0; o >>= 1) {
    if (t < o) { smn[t] = fminf(smn[t], smn[t + o]); smx[t] = fmaxf(smx[t], smx[t + o]); }
    __syncthreads();
  }
  if (t == 0) { wmn[b] = smn[0]; wmx[b] = smx[0]; }
}

// ---------------------------------------------------------------------------
// Kernel 4: LayerNorm(cat2) + x*w + xi*(1-w); one (b,l) row per block.
// ---------------------------------------------------------------------------
__global__ void __launch_bounds__(256) k4_out(const float* __restrict__ x,
                                              const float* __restrict__ xi,
                                              const float* __restrict__ sc,
                                              const float* __restrict__ W,
                                              const float* __restrict__ wmn,
                                              const float* __restrict__ wmx,
                                              const int* __restrict__ pos,
                                              const float* __restrict__ lnw,
                                              const float* __restrict__ lnb,
                                              float* __restrict__ out) {
  __shared__ float ssum[256], ssq[256];
  const int row = blockIdx.x;         // b*1152 + l
  const int b = row / L_;
  const int l = row - b * L_;
  const int d = threadIdx.x;
  const size_t base = (size_t)row * D_;

  float v;
  if (l < TF_) {
    v = x[base + d];
  } else {
    const int s = l - TF_;
    const float mn = wmn[b], mx = wmx[b];
    const float wn = (W[b * S_ + s] - mn) / (mx - mn);
    v = sc[((size_t)b * S_ + s) * D_ + d] * wn;
  }
  ssum[d] = v; ssq[d] = v * v;
  __syncthreads();
  for (int o = 128; o > 0; o >>= 1) {
    if (d < o) { ssum[d] += ssum[d + o]; ssq[d] += ssq[d + o]; }
    __syncthreads();
  }
  const float mu  = ssum[0] * (1.0f / D_);
  const float var = ssq[0] * (1.0f / D_) - mu * mu;
  const float ln  = (v - mu) * rsqrtf(var + 1e-5f) * lnw[d] + lnb[d];
  const float wp  = (float)pos[b] * (1.0f / ((float)TF_ * (float)S_));
  out[base + d] = x[base + d] * wp + xi[base + d] * (1.0f - wp) + ln;
}

extern "C" void kernel_launch(void* const* d_in, const int* in_sizes, int n_in,
                              void* d_out, int out_size, void* d_ws, size_t ws_size,
                              hipStream_t stream) {
  (void)in_sizes; (void)n_in; (void)out_size; (void)ws_size;
  const float* x   = (const float*)d_in[0];
  const float* xi  = (const float*)d_in[1];
  const float* ft  = (const float*)d_in[2];
  const float* cw  = (const float*)d_in[3];
  const float* cb  = (const float*)d_in[4];
  const float* lnw = (const float*)d_in[5];
  const float* lnb = (const float*)d_in[6];
  float* out = (float*)d_out;

  char*  ws  = (char*)d_ws;
  float* sc  = (float*)(ws);                                   // 67,108,864 B
  float* W   = (float*)(ws + 67108864);                        //    262,144 B
  float* fm  = (float*)(ws + 67108864 + 262144);               //     65,536 B
  float* wmn = (float*)(ws + 67108864 + 262144 + 65536);
  float* wmx = wmn + B_;
  int*   pos = (int*)(wmx + B_);

  k0_init   <<<1,    64,  0, stream>>>(pos);
  k1_count  <<<4096, 256, 0, stream>>>(x, xi, ft, pos);
  k2_conv   <<<4096, 256, 0, stream>>>(x, xi, cw, cb, sc);
  k3a_ftmean<<<B_,   256, 0, stream>>>(ft, fm);
  k3b_W     <<<256,  256, 0, stream>>>(sc, fm, W);
  k3c_minmax<<<B_,   256, 0, stream>>>(W, wmn, wmx);
  k4_out    <<<B_ * L_, 256, 0, stream>>>(x, xi, sc, W, wmn, wmx, pos, lnw, lnb, out);
}
